// DeformAttn_15015205666977
// MI455X (gfx1250) — matrix operand order repaired and checked by reading, not compiled, not run on hardware
//
#include <hip/hip_runtime.h>
#include <hip/hip_bf16.h>
#include <math.h>

// ---------------------------------------------------------------------------
// Deformable attention for MI455X (gfx1250), wave32 + WMMA.
//
// Shapes: B=4, N=4096, D=256, NH=8, NP=8, H=W=256, HD=32, M=B*NH=32.
//
// Roofline: ~15 GFLOP vs ~630 MB HBM traffic @ 23.3 TB/s -> memory bound.
//   * bev_feat read (256 MB) + channel-last f16 value write (128 MB) +
//     coalesced 64B-per-corner gathers (<=256 MB) dominate.
//   * All GEMMs use v_wmma_f32_16x16x32_f16; weights pre-transposed and
//     pre-converted to f16 [N][K] so every B fragment is 32 contiguous bytes
//     per lane (2x global_load_b128, zero cvt in the hot loop).
//   * Hidden activations kept f16 end-to-end; f32 only where precision
//     matters (sampling positions, final output).
// ---------------------------------------------------------------------------

typedef __attribute__((ext_vector_type(16))) _Float16 v16h;
typedef __attribute__((ext_vector_type(8)))  float    v8f;

#define HW_   (256 * 256)
#define MROWS 16384          // B*N
#define KDIM  256            // D

// ---------------------------------------------------------------------------
// One-shot weight transpose + f32->f16 convert:  w[K][N] -> wt[N][K]
// ---------------------------------------------------------------------------
__global__ void wt_convert_kernel(const float* __restrict__ w,
                                  _Float16* __restrict__ wt, int K, int N) {
    const int idx = blockIdx.x * blockDim.x + threadIdx.x;
    if (idx >= K * N) return;
    const int k = idx / N;
    const int n = idx - k * N;
    wt[(size_t)n * K + k] = (_Float16)w[idx];
}

// ---------------------------------------------------------------------------
// C[M,N] = act(A[M,K] @ W[K,N] + bias[N]) with W given as Wt[N][K] f16.
// One wave computes a 16x64 tile: one A fragment feeds 4 WMMAs.
//
// WMMA fragment layouts used here (contiguous k, not the ISA table order):
//   A (16x32 f16): lane<16 -> row=lane,   K in {0..15}
//                  lane>=16 -> row=lane-16, K in {16..31}
//   B (32x16 f16): col = lane&15; lane<16 -> K=0..15, else K=16..31
//                  -> 16 contiguous f16 from Wt[col][k0 + (lane&16?16:0) ..]
//   D: lane = n + 16*(m>=8); element j -> row (lane>>4)*8 + j
// ---------------------------------------------------------------------------
template <typename AT, typename OT>
__global__ void gemm_bias_act_kernel(const AT* __restrict__ A,
                                     const _Float16* __restrict__ Wt,
                                     const float* __restrict__ bias,
                                     OT* __restrict__ C,
                                     int M, int N, int K, int relu) {
    const int lane = threadIdx.x & 31;
    const int wid  = (blockIdx.x * blockDim.x + threadIdx.x) >> 5;
    const int ntiles = N >> 6;                    // 64-wide tiles
    const int tn = wid % ntiles;
    const int tm = wid / ntiles;
    if (tm * 16 >= M) return;

    const int row   = tm * 16 + (lane & 15);
    const int n0    = tn * 64;
    const int abase = (lane & 16) ? 16 : 0;
    const int bofs  = (lane & 16) ? 16 : 0;

    v8f acc[4] = {v8f{}, v8f{}, v8f{}, v8f{}};

    for (int k0 = 0; k0 < K; k0 += 32) {
        v16h a;
        const AT* arow = A + (size_t)row * K + k0;
#pragma unroll
        for (int i = 0; i < 8; ++i)  a[i] = (_Float16)arow[abase + i];
#pragma unroll
        for (int i = 8; i < 16; ++i) a[i] = (_Float16)arow[abase + i];

#pragma unroll
        for (int t = 0; t < 4; ++t) {
            const _Float16* brow =
                Wt + (size_t)(n0 + t * 16 + (lane & 15)) * K + k0 + bofs;
            v16h b;
#pragma unroll
            for (int i = 0; i < 16; ++i) b[i] = brow[i];   // 32B contiguous
            acc[t] = __builtin_amdgcn_wmma_f32_16x16x32_f16(
                false, a, false, b, (short)0, acc[t], false, false);
        }
    }

#pragma unroll
    for (int t = 0; t < 4; ++t) {
        const int col = n0 + t * 16 + (lane & 15);
        const float bv = bias[col];
#pragma unroll
        for (int j = 0; j < 8; ++j) {
            const int m = ((lane >> 4) << 3) + j;
            float v = acc[t][j] + bv;
            if (relu) v = fmaxf(v, 0.0f);
            C[(size_t)(tm * 16 + m) * N + col] = (OT)v;
        }
    }
}

// ---------------------------------------------------------------------------
// Value projection + layout transpose:
//   in : bev [m=32][c=32][HW]  (f32, channel-major)
//   out: val [m=32][HW][o=32]  (f16, channel-LAST for coalesced gathers)
// out[p,o] = sum_c in[c,p]*vpw[o,c] + vpb[o].  K=32 == one WMMA.
// Each wave processes 4 consecutive 16-pixel tiles (8 WMMAs), amortizing
// the hoisted vp_w B-fragments.
// ---------------------------------------------------------------------------
__global__ void value_proj_kernel(const float* __restrict__ bev,
                                  const float* __restrict__ vpw,   // [o=32][c=32]
                                  const float* __restrict__ vpb,   // [32]
                                  _Float16* __restrict__ out) {
    const int lane = threadIdx.x & 31;
    const int wid  = (blockIdx.x * blockDim.x + threadIdx.x) >> 5;
    const int grp  = wid & 1023;    // 1024 groups of 64 pixels
    const int m    = wid >> 10;     // head-group 0..31
    const int pg   = grp * 64;

    const float* src = bev + (size_t)m * 32 * HW_;

    // B fragments: B[k=c][n=o] = vpw[o][c]; constant across the pixel loop.
    v16h b0, b1;
    {
        const int n  = lane & 15;
        const int kb = (lane & 16) ? 16 : 0;
#pragma unroll
        for (int i = 0; i < 16; ++i) {
            const int c = kb + i;
            b0[i] = (_Float16)vpw[n * 32 + c];
            b1[i] = (_Float16)vpw[(n + 16) * 32 + c];
        }
    }

    const int o     = lane & 15;
    const float bv0 = vpb[o];
    const float bv1 = vpb[o + 16];
    const int abase = (lane & 16) ? 16 : 0;

#pragma unroll
    for (int t = 0; t < 4; ++t) {
        const int p0 = pg + t * 16;
        const int p  = p0 + (lane & 15);

        v16h a;   // rows = pixels, K = channels (lanes 0..15 coalesce 64B/chan)
#pragma unroll
        for (int i = 0; i < 8; ++i)
            a[i] = (_Float16)src[(size_t)(abase + i) * HW_ + p];
#pragma unroll
        for (int i = 8; i < 16; ++i)
            a[i] = (_Float16)src[(size_t)(abase + i) * HW_ + p];

        v8f acc0 = {}, acc1 = {};
        acc0 = __builtin_amdgcn_wmma_f32_16x16x32_f16(false, a, false, b0, (short)0, acc0, false, false);
        acc1 = __builtin_amdgcn_wmma_f32_16x16x32_f16(false, a, false, b1, (short)0, acc1, false, false);

#pragma unroll
        for (int j = 0; j < 8; ++j) {
            const int prow = ((lane >> 4) << 3) + j;
            const size_t base = ((size_t)m * HW_ + (size_t)(p0 + prow)) * 32;
            out[base + o]      = (_Float16)(acc0[j] + bv0);
            out[base + o + 16] = (_Float16)(acc1[j] + bv1);
        }
    }
}

// ---------------------------------------------------------------------------
// Grid-sample (bilinear, zero pad, y flipped) + softmax(aw) + point-weighted
// sum. One wave per (b, n, head); lane = channel (HD=32). Each corner fetch
// is 32 lanes * 2B = 64 contiguous bytes. Branchless: clamped indices +
// validity multipliers (value tensor is fully initialized -> safe).
// ---------------------------------------------------------------------------
__global__ void sample_kernel(const float* __restrict__ ref_pos,  // [B*N][2]
                              const float* __restrict__ so,       // [B*N][128]
                              const float* __restrict__ aw,       // [B*N][64]
                              const _Float16* __restrict__ value, // [32][HW][32]
                              _Float16* __restrict__ attn_out) {  // [B*N][256]
    const int lane = threadIdx.x & 31;
    const int wid  = (blockIdx.x * blockDim.x + threadIdx.x) >> 5;
    const int h  = wid & 7;
    const int bn = wid >> 3;          // 0..16383
    const int b  = bn >> 12;          // N = 4096
    const int m  = b * 8 + h;

    const float rx = ref_pos[bn * 2 + 0];
    const float ry = ref_pos[bn * 2 + 1];
    const float* sop = so + (size_t)bn * 128 + h * 16;  // NP*2 per head
    const float* awp = aw + (size_t)bn * 64 + h * 8;

    // softmax over NP=8 (broadcast loads; redundant per lane but cheap)
    float av[8];
    float wmax = -INFINITY;
#pragma unroll
    for (int p = 0; p < 8; ++p) { av[p] = awp[p]; wmax = fmaxf(wmax, av[p]); }
    float s = 0.0f;
#pragma unroll
    for (int p = 0; p < 8; ++p) { av[p] = __expf(av[p] - wmax); s += av[p]; }
    const float inv = 1.0f / s;

    const _Float16* vbase = value + (size_t)m * HW_ * 32;
    const int c = lane;
    float acc = 0.0f;

#pragma unroll
    for (int p = 0; p < 8; ++p) {
        const float gx = rx + sop[p * 2 + 0] * (1.0f / 256.0f);
        const float gy = -(ry + sop[p * 2 + 1] * (1.0f / 256.0f));
        const float x = (gx + 1.0f) * 128.0f - 0.5f;
        const float y = (gy + 1.0f) * 128.0f - 0.5f;
        const float xf = floorf(x), yf = floorf(y);
        const int x0 = (int)xf, y0 = (int)yf;
        const float wx1 = x - xf, wy1 = y - yf;
        const float wx0 = 1.0f - wx1, wy0 = 1.0f - wy1;

        const float vx0 = (x0 >= 0 && x0 <= 255) ? 1.0f : 0.0f;
        const float vx1 = (x0 + 1 >= 0 && x0 + 1 <= 255) ? 1.0f : 0.0f;
        const float vy0 = (y0 >= 0 && y0 <= 255) ? 1.0f : 0.0f;
        const float vy1 = (y0 + 1 >= 0 && y0 + 1 <= 255) ? 1.0f : 0.0f;

        const int xc0 = min(max(x0, 0), 255);
        const int xc1 = min(max(x0 + 1, 0), 255);
        const int yc0 = min(max(y0, 0), 255);
        const int yc1 = min(max(y0 + 1, 0), 255);

        const _Float16* r0 = vbase + ((size_t)yc0 * 256) * 32;
        const _Float16* r1 = vbase + ((size_t)yc1 * 256) * 32;
        const float v00 = (float)r0[xc0 * 32 + c];
        const float v01 = (float)r0[xc1 * 32 + c];
        const float v10 = (float)r1[xc0 * 32 + c];
        const float v11 = (float)r1[xc1 * 32 + c];

        const float samp = v00 * (wy0 * wx0 * vy0 * vx0)
                         + v01 * (wy0 * wx1 * vy0 * vx1)
                         + v10 * (wy1 * wx0 * vy1 * vx0)
                         + v11 * (wy1 * wx1 * vy1 * vx1);
        acc += av[p] * inv * samp;
    }

    attn_out[(size_t)bn * 256 + h * 32 + c] = (_Float16)acc;
}

// ---------------------------------------------------------------------------
extern "C" void kernel_launch(void* const* d_in, const int* in_sizes, int n_in,
                              void* d_out, int out_size, void* d_ws, size_t ws_size,
                              hipStream_t stream) {
    const float* q      = (const float*)d_in[0];   // ba_query [B,N,D]
    const float* refp   = (const float*)d_in[1];   // ref_pos  [B,N,2]
    const float* bev    = (const float*)d_in[2];   // bev_feat [B,D,H,W]
    const float* so_w1  = (const float*)d_in[3];
    const float* so_b1  = (const float*)d_in[4];
    const float* so_w2  = (const float*)d_in[5];
    const float* so_b2  = (const float*)d_in[6];
    const float* aw_w1  = (const float*)d_in[7];
    const float* aw_b1  = (const float*)d_in[8];
    const float* aw_w2  = (const float*)d_in[9];
    const float* aw_b2  = (const float*)d_in[10];
    const float* vp_w   = (const float*)d_in[11];
    const float* vp_b   = (const float*)d_in[12];
    const float* op_w1  = (const float*)d_in[13];
    const float* op_b1  = (const float*)d_in[14];
    const float* op_w2  = (const float*)d_in[15];
    const float* op_b2  = (const float*)d_in[16];
    float* out = (float*)d_out;

    // Workspace layout (~157 MB)
    char* ws = (char*)d_ws;
    size_t off = 0;
    _Float16* value_t = (_Float16*)(ws + off); off += (size_t)32 * HW_ * 32 * 2; // 128 MB
    _Float16* hiddenh = (_Float16*)(ws + off); off += (size_t)MROWS * 256 * 2;   // 8 MB
    _Float16* attnh   = (_Float16*)(ws + off); off += (size_t)MROWS * 256 * 2;   // 8 MB
    float*    so_buf  = (float*)(ws + off);    off += (size_t)MROWS * 128 * 4;   // 8 MB
    float*    aw_buf  = (float*)(ws + off);    off += (size_t)MROWS * 64  * 4;   // 4 MB
    _Float16* so_w1t  = (_Float16*)(ws + off); off += (size_t)256 * 256 * 2;
    _Float16* so_w2t  = (_Float16*)(ws + off); off += (size_t)256 * 128 * 2;
    _Float16* aw_w1t  = (_Float16*)(ws + off); off += (size_t)256 * 256 * 2;
    _Float16* aw_w2t  = (_Float16*)(ws + off); off += (size_t)256 * 64  * 2;
    _Float16* op_w1t  = (_Float16*)(ws + off); off += (size_t)256 * 256 * 2;
    _Float16* op_w2t  = (_Float16*)(ws + off); off += (size_t)256 * 256 * 2;

    // ---- one-shot weight transpose/convert (tiny) ----
    auto wconv = [&](const float* w, _Float16* wt, int K, int N) {
        const int n = K * N;
        wt_convert_kernel<<<(n + 255) / 256, 256, 0, stream>>>(w, wt, K, N);
    };
    wconv(so_w1, so_w1t, KDIM, 256);
    wconv(so_w2, so_w2t, KDIM, 128);
    wconv(aw_w1, aw_w1t, KDIM, 256);
    wconv(aw_w2, aw_w2t, KDIM, 64);
    wconv(op_w1, op_w1t, KDIM, 256);
    wconv(op_w2, op_w2t, KDIM, 256);

    // ---- GEMM launcher: 16x64 tile per wave, 8 waves / 256-thread block ----
    auto blocks_for = [](int N) { return ((MROWS / 16) * (N / 64)) / 8; };

    // Sampling-offset MLP: so = relu(q@so_w1+b1) @ so_w2 + b2   [16384,128]
    gemm_bias_act_kernel<float, _Float16>
        <<<blocks_for(256), 256, 0, stream>>>(q, so_w1t, so_b1, hiddenh,
                                              MROWS, 256, KDIM, 1);
    gemm_bias_act_kernel<_Float16, float>
        <<<blocks_for(128), 256, 0, stream>>>(hiddenh, so_w2t, so_b2, so_buf,
                                              MROWS, 128, KDIM, 0);
    // Attention-weight MLP: aw = relu(q@aw_w1+b1) @ aw_w2 + b2  [16384,64]
    gemm_bias_act_kernel<float, _Float16>
        <<<blocks_for(256), 256, 0, stream>>>(q, aw_w1t, aw_b1, hiddenh,
                                              MROWS, 256, KDIM, 1);
    gemm_bias_act_kernel<_Float16, float>
        <<<blocks_for(64), 256, 0, stream>>>(hiddenh, aw_w2t, aw_b2, aw_buf,
                                             MROWS, 64, KDIM, 0);

    // Value projection + transpose to channel-last f16. 32768 waves.
    value_proj_kernel<<<4096, 256, 0, stream>>>(bev, vp_w, vp_b, value_t);

    // Bilinear sampling + softmax + point reduction. 131072 waves.
    sample_kernel<<<16384, 256, 0, stream>>>(refp, so_buf, aw_buf, value_t, attnh);

    // Output MLP -> d_out
    gemm_bias_act_kernel<_Float16, _Float16>
        <<<blocks_for(256), 256, 0, stream>>>(attnh, op_w1t, op_b1, hiddenh,
                                              MROWS, 256, KDIM, 1);
    gemm_bias_act_kernel<_Float16, float>
        <<<blocks_for(256), 256, 0, stream>>>(hiddenh, op_w2t, op_b2, out,
                                              MROWS, 256, KDIM, 0);
}
